// EIRNNLayers_53455162966515
// MI455X (gfx1250) — matrix operand order repaired and checked by reading, not compiled, not run on hardware
//
#include <hip/hip_runtime.h>
#include <hip/hip_bf16.h>

// ---------------- problem constants ----------------
#define T_STEPS   128
#define BATCH     64
#define INDIM     512
#define HDIM      1024
#define NLAYERS   3
#define NE        819          // int(1024 * 0.8)
#define ALPHA     0.2f
#define NWG       16           // 16 workgroups x 64 output units = 1024
#define LDS_STRIDE 1048        // bf16 elems per activation row (2096B: 16B aligned, bank-spread)
#define SMEM_BYTES (BATCH * LDS_STRIDE * 2)

typedef __attribute__((ext_vector_type(16))) __bf16 v16bf;
typedef __attribute__((ext_vector_type(8)))  float  v8f;

union FragU { v16bf v; uint4 u[2]; };
union Pack4 { __bf16 h[4]; uint2 u; };

// ---------------- WMMA wrapper ----------------
__device__ __forceinline__ v8f wmma_bf16(v16bf a, v16bf b, v8f c) {
    return __builtin_amdgcn_wmma_f32_16x16x32_bf16(
        /*neg_a=*/false, a, /*neg_b=*/false, b,
        /*c_mod=*/(short)0, c, /*reuse_a=*/false, /*reuse_b=*/false);
}

// A fragment (16x32 bf16) from LDS: lane&15 = row m; lanes<16 hold K {0..7,16..23},
// lanes>=16 hold K {8..15,24..31}  (ISA 16-bit A layout).
__device__ __forceinline__ v16bf load_a_frag(const __bf16* __restrict__ lds,
                                             int mbase, int kb, int lane) {
    int m = mbase + (lane & 15);
    const __bf16* p = lds + m * LDS_STRIDE + kb + ((lane >> 4) << 3);
    FragU f;
    f.u[0] = *(const uint4*)(p);
    f.u[1] = *(const uint4*)(p + 16);
    return f.v;
}

// GEMM over K: acc{0,1} += A(lds) @ W^T.  KD compile-time; unroll bounded to 2
// so only two iterations of fragments are live (stays below the spill cliff).
template<int KD>
__device__ __forceinline__ void gemm_accum(const __bf16* __restrict__ W,
                                           const __bf16* __restrict__ lds,
                                           int r, int lane, int mbase0, int mbase1,
                                           v8f& acc0, v8f& acc1) {
    const __bf16* wrow = W + (size_t)r * KD + ((lane >> 4) << 3);
    #pragma unroll 2
    for (int kb = 0; kb < KD; kb += 32) {
        FragU fb;
        fb.u[0] = *(const uint4*)(wrow + kb);
        fb.u[1] = *(const uint4*)(wrow + kb + 16);
        v16bf a0 = load_a_frag(lds, mbase0, kb, lane);
        v16bf a1 = load_a_frag(lds, mbase1, kb, lane);
        acc0 = wmma_bf16(a0, fb.v, acc0);
        acc1 = wmma_bf16(a1, fb.v, acc1);
    }
}

// ---- staging: bf16 [B,H] global -> padded LDS.  4 threads/row, 32 uint4 each,
// unroll 8 => batches of 8 loads in flight before the LDS stores. ----
__device__ __forceinline__ void stage_h(const __bf16* __restrict__ src,
                                        __bf16* __restrict__ alds, int tid) {
    int row = tid >> 2;                  // 64 rows
    int seg = tid & 3;                   // 4 x 32 uint4 (512B) segments per row
    const uint4* s = (const uint4*)(src + (size_t)row * HDIM) + seg * 32;
    uint4* d = (uint4*)((char*)alds + (size_t)row * (LDS_STRIDE * 2)) + seg * 32;
    #pragma unroll 8
    for (int i = 0; i < 32; ++i) d[i] = s[i];
}

// ---- staging: f32 [B,IN] global -> bf16 padded LDS (layer-0 input). ----
__device__ __forceinline__ void stage_x(const float* __restrict__ xt,
                                        __bf16* __restrict__ alds, int tid) {
    int row = tid >> 2;                  // 64 rows
    int seg = tid & 3;                   // 4 x 32 float4 segments per row
    const float4* s = (const float4*)(xt + (size_t)row * INDIM) + seg * 32;
    uint2* d = (uint2*)((char*)alds + (size_t)row * (LDS_STRIDE * 2)) + seg * 32;
    #pragma unroll 4
    for (int i = 0; i < 32; ++i) {
        float4 f = s[i];
        Pack4 p;
        p.h[0] = (__bf16)f.x; p.h[1] = (__bf16)f.y;
        p.h[2] = (__bf16)f.z; p.h[3] = (__bf16)f.w;
        d[i] = p.u;
    }
}

// ---------------- device-wide barrier (16 co-resident WGs) ----------------
__device__ __forceinline__ void grid_barrier(unsigned* bar, unsigned* epoch) {
    __threadfence();           // release this WG's global stores
    __syncthreads();
    if (threadIdx.x == 0) {
        *epoch += NWG;
        __hip_atomic_fetch_add(bar, 1u, __ATOMIC_RELEASE, __HIP_MEMORY_SCOPE_AGENT);
        while (__hip_atomic_load(bar, __ATOMIC_ACQUIRE, __HIP_MEMORY_SCOPE_AGENT) < *epoch) {
            __builtin_amdgcn_s_sleep(2);
        }
    }
    __syncthreads();
    __threadfence();           // acquire other WGs' stores
}

// ---------------- one layer of one timestep, fully specialized ----------------
template<int J, int KD>
__device__ __forceinline__ void layer_step(
        int t, int tid, int lane, int r, int mbase0, int mbase1, int half,
        float emask_r,
        const float* __restrict__ inputs,
        const float* __restrict__ bvec,
        const __bf16* __restrict__ Wi,       // [H][KD] bf16 |W_in|
        const __bf16* __restrict__ Wr,       // [H][H]  bf16 Dale-signed |W_rec|
        __bf16* __restrict__ hprev,          // [L][2][B][H]
        __bf16* __restrict__ hmask,          // [2][B][H]
        float*  __restrict__ outputs,
        float*  __restrict__ all_hiddens,
        __bf16* __restrict__ alds,
        float (&vst)[2][8]) {
    const size_t nBH = (size_t)BATCH * HDIM;

    // ---- stage layer input into LDS as bf16 ----
    if (J == 0) stage_x(inputs + (size_t)t * BATCH * INDIM, alds, tid);
    else        stage_h(hmask + (size_t)(J - 1) * nBH, alds, tid);
    __syncthreads();

    v8f acc0 = {0.f, 0.f, 0.f, 0.f, 0.f, 0.f, 0.f, 0.f};
    v8f acc1 = {0.f, 0.f, 0.f, 0.f, 0.f, 0.f, 0.f, 0.f};

    // ---- feedforward GEMM: acc += x @ Win^T ----
    gemm_accum<KD>(Wi, alds, r, lane, mbase0, mbase1, acc0, acc1);
    __syncthreads();

    // ---- stage h_prev[J] (parity t&1) and recurrent GEMM ----
    stage_h(hprev + ((size_t)J * 2 + (t & 1)) * nBH, alds, tid);
    __syncthreads();
    gemm_accum<HDIM>(Wr, alds, r, lane, mbase0, mbase1, acc0, acc1);

    // ---- leaky update, relu, stores ----
    const float bias = bvec[J * HDIM + r];
    __bf16* hpw = hprev + ((size_t)J * 2 + ((t + 1) & 1)) * nBH;
    float* ah = all_hiddens + ((size_t)J * (T_STEPS + 1) + (t + 1)) * nBH;

    #pragma unroll
    for (int p = 0; p < 2; ++p) {
        v8f acc = p ? acc1 : acc0;
        int mb  = p ? mbase1 : mbase0;
        #pragma unroll
        for (int i = 0; i < 8; ++i) {
            int m = mb + i + half * 8;                 // C layout: VGPR i, lane half
            float pre = acc[i] + bias;
            float v = (1.0f - ALPHA) * vst[p][i] + ALPHA * pre;
            vst[p][i] = v;
            float h = v > 0.0f ? v : 0.0f;
            ah[(size_t)m * HDIM + r] = h;
            hpw[(size_t)m * HDIM + r] = (__bf16)h;
            float hm = h * emask_r;
            if (J < 2)
                hmask[(size_t)J * nBH + (size_t)m * HDIM + r] = (__bf16)hm;
            else
                outputs[((size_t)t * BATCH + m) * HDIM + r] = hm;
        }
    }
}

// ---------------- init: weight bf16 conversion, state init, barrier reset ----------------
__global__ void ei_init_kernel(const float* __restrict__ Win0,
                               const float* __restrict__ Win1,
                               const float* __restrict__ Win2,
                               const float* __restrict__ Wrec,
                               const float* __restrict__ h0,   // hidden_init[0] : [B,H]
                               __bf16* __restrict__ Wb0, __bf16* __restrict__ Wb1,
                               __bf16* __restrict__ Wb2, __bf16* __restrict__ Wrb,
                               __bf16* __restrict__ hprev,     // [L][2][B][H]
                               float*  __restrict__ all_hiddens,
                               unsigned* __restrict__ bar) {
    size_t idx    = (size_t)blockIdx.x * blockDim.x + threadIdx.x;
    size_t stride = (size_t)gridDim.x * blockDim.x;
    if (idx == 0) *bar = 0u;

    const size_t nW0 = (size_t)HDIM * INDIM;
    const size_t nW  = (size_t)HDIM * HDIM;
    for (size_t i = idx; i < nW0; i += stride) Wb0[i] = (__bf16)fabsf(Win0[i]);
    for (size_t i = idx; i < nW;  i += stride) Wb1[i] = (__bf16)fabsf(Win1[i]);
    for (size_t i = idx; i < nW;  i += stride) Wb2[i] = (__bf16)fabsf(Win2[i]);
    for (size_t i = idx; i < NLAYERS * nW; i += stride) {
        int k = (int)(i & (HDIM - 1));                  // presynaptic column -> Dale sign
        float s = (k < NE) ? 1.0f : -1.0f;
        Wrb[i] = (__bf16)(fabsf(Wrec[i]) * s);
    }
    const size_t nBH = (size_t)BATCH * HDIM;
    for (size_t i = idx; i < nBH; i += stride) {
        float h = h0[i];
        __bf16 hb = (__bf16)h;
        for (int j = 0; j < NLAYERS; ++j) {
            hprev[((size_t)j * 2 + 0) * nBH + i] = hb;                     // parity 0
            all_hiddens[(size_t)j * (T_STEPS + 1) * nBH + i] = h;          // t = 0 slice
        }
    }
}

// ---------------- persistent EI-RNN kernel ----------------
__global__ void __launch_bounds__(256, 1)
ei_rnn_persistent(const float* __restrict__ inputs,     // [T,B,IN]
                  const float* __restrict__ bvec,       // [L,H]
                  const __bf16* __restrict__ Wb0,
                  const __bf16* __restrict__ Wb1,
                  const __bf16* __restrict__ Wb2,
                  const __bf16* __restrict__ Wrb,       // [L,H,H]
                  __bf16* __restrict__ hprev,           // [L][2][B][H]
                  __bf16* __restrict__ hmask,           // [2][B][H]
                  float*  __restrict__ outputs,         // [T,B,H]
                  float*  __restrict__ all_hiddens,     // [L,T+1,B,H]
                  unsigned* __restrict__ bar) {
    extern __shared__ __bf16 alds[];                    // [BATCH][LDS_STRIDE]

    const int tid  = threadIdx.x;
    const int lane = tid & 31;
    const int wave = tid >> 5;
    const int cb   = wave & 3;                // N-block shared by both tiles of this wave
    const int rb0  = wave >> 2;               // M-blocks rb0 and rb0+2
    const int nbase  = cb * 16;
    const int mbase0 = rb0 * 16;
    const int mbase1 = (rb0 + 2) * 16;
    const int half   = lane >> 4;

    const int r = blockIdx.x * 64 + nbase + (lane & 15);   // global output unit
    const float emask_r = (r < NE) ? 1.0f : 0.0f;

    float vst[NLAYERS][2][8];
    #pragma unroll
    for (int j = 0; j < NLAYERS; ++j)
        #pragma unroll
        for (int p = 0; p < 2; ++p)
            #pragma unroll
            for (int i = 0; i < 8; ++i) vst[j][p][i] = 0.0f;

    unsigned epoch = 0;

    #pragma unroll 1
    for (int t = 0; t < T_STEPS; ++t) {
        layer_step<0, INDIM>(t, tid, lane, r, mbase0, mbase1, half, emask_r,
                             inputs, bvec, Wb0, Wrb + 0 * (size_t)HDIM * HDIM,
                             hprev, hmask, outputs, all_hiddens, alds, vst[0]);
        grid_barrier(bar, &epoch);

        layer_step<1, HDIM>(t, tid, lane, r, mbase0, mbase1, half, emask_r,
                            inputs, bvec, Wb1, Wrb + 1 * (size_t)HDIM * HDIM,
                            hprev, hmask, outputs, all_hiddens, alds, vst[1]);
        grid_barrier(bar, &epoch);

        layer_step<2, HDIM>(t, tid, lane, r, mbase0, mbase1, half, emask_r,
                            inputs, bvec, Wb2, Wrb + 2 * (size_t)HDIM * HDIM,
                            hprev, hmask, outputs, all_hiddens, alds, vst[2]);
        grid_barrier(bar, &epoch);
    }
}

// ---------------- host launch ----------------
extern "C" void kernel_launch(void* const* d_in, const int* in_sizes, int n_in,
                              void* d_out, int out_size, void* d_ws, size_t ws_size,
                              hipStream_t stream) {
    const float* inputs      = (const float*)d_in[0];   // [128,64,512]
    const float* hidden_init = (const float*)d_in[1];   // [1,64,1024]
    const float* W_in0       = (const float*)d_in[2];   // [1024,512]
    const float* W_in1       = (const float*)d_in[3];   // [1024,1024]
    const float* W_in2       = (const float*)d_in[4];   // [1024,1024]
    const float* W_rec       = (const float*)d_in[5];   // [3,1024,1024]
    const float* bvec        = (const float*)d_in[6];   // [3,1024]

    float* outputs     = (float*)d_out;                                   // [128,64,1024]
    float* all_hiddens = outputs + (size_t)T_STEPS * BATCH * HDIM;        // [3,129,64,1024]

    char* ws = (char*)d_ws;
    __bf16* Wb0 = (__bf16*)ws;  ws += (size_t)HDIM * INDIM * 2;
    __bf16* Wb1 = (__bf16*)ws;  ws += (size_t)HDIM * HDIM * 2;
    __bf16* Wb2 = (__bf16*)ws;  ws += (size_t)HDIM * HDIM * 2;
    __bf16* Wrb = (__bf16*)ws;  ws += (size_t)NLAYERS * HDIM * HDIM * 2;
    __bf16* hprev = (__bf16*)ws; ws += (size_t)NLAYERS * 2 * BATCH * HDIM * 2;
    __bf16* hmask = (__bf16*)ws; ws += (size_t)2 * BATCH * HDIM * 2;
    unsigned* bar = (unsigned*)ws;

    hipLaunchKernelGGL(ei_init_kernel, dim3(2048), dim3(256), 0, stream,
                       W_in0, W_in1, W_in2, W_rec, hidden_init,
                       Wb0, Wb1, Wb2, Wrb, hprev, all_hiddens, bar);

    hipLaunchKernelGGL(ei_rnn_persistent, dim3(NWG), dim3(256), SMEM_BYTES, stream,
                       inputs, bvec, Wb0, Wb1, Wb2, Wrb,
                       hprev, hmask, outputs, all_hiddens, bar);
}
